// RuleGraphNet_83958020702806
// MI455X (gfx1250) — matrix operand order repaired and checked by reading, not compiled, not run on hardware
//
#include <hip/hip_runtime.h>
#include <hip/hip_bf16.h>
#include <stdint.h>

// ---------------------------------------------------------------------------
// CDNA5 (gfx1250) WMMA GNN: two TripleConv edge-GEMMs + two node MLPs.
// v_wmma_f32_16x16x32_f16 everywhere; weights pre-packed to f16 in the exact
// B-operand VGPR layout, staged into LDS with b128 copies, read back with
// ds_load_b128.  A-side: scattered per-edge concat gather (f32 -> f16).
// ---------------------------------------------------------------------------

typedef __attribute__((ext_vector_type(16))) _Float16 v16h;
typedef __attribute__((ext_vector_type(8)))  float    v8f;

#define NNODES 50000
#define NEDGES 800000
#define D1 107      // node dim into conv1 (and out of conv1)
#define ED 100      // edge dim
#define H1D 100     // hidden_dim1
#define H2D 64      // hidden_dim2
#define DIMO 100    // output dim

// ---------------------------------------------------------------------------
// Pack W[K x Nout] (f32, row-major) into f16 tiles matching the CDNA5 WMMA
// B-matrix layout.
//   P[(((nt*KC)+kc)*32 + lane)*16 + j] = W[k*Nout + n]  (0 if out of range)
//   n = nt*16 + (lane&15),  k = kc*32 + (lane<16 ? 0 : 8) + (j&7) + (j>=8 ? 16 : 0)
// ---------------------------------------------------------------------------
__global__ void pack_weight_f16(const float* __restrict__ W,
                                _Float16* __restrict__ P,
                                int K, int Nout, int KC, int NT) {
  int idx = blockIdx.x * blockDim.x + threadIdx.x;
  int total = KC * NT * 32 * 16;
  if (idx >= total) return;
  int j    = idx & 15;
  int lane = (idx >> 4) & 31;
  int t    = idx >> 9;
  int kc   = t % KC;
  int nt   = t / KC;
  int n = nt * 16 + (lane & 15);
  int k = kc * 32 + ((lane < 16) ? 0 : 8) + (j & 7) + ((j >> 3) << 4);
  P[idx] = (_Float16)((n < Nout && k < K) ? W[(long)k * Nout + n] : 0.0f);
}

// Gathered concat feature for edge GEMM A-matrix.
// first half edges : [x_dst | e | x_src]  (xi, edge_attr, xj)
// second half edges: [x_src | e | x_dst]  (xj, edge_attr, xi)
__device__ __forceinline__ float fetch_concat(const float* __restrict__ x,
                                              const float* __restrict__ ea,
                                              int D, int Kin,
                                              int se, int de, long e,
                                              int k, bool second) {
  if (k >= Kin) return 0.0f;
  if (k < D)      return x[(long)(second ? se : de) * D + k];
  if (k < D + ED) return ea[e * ED + (k - D)];
  return x[(long)(second ? de : se) * D + (k - D - ED)];
}

// b128 LDS staging of a packed panel (HALVES must be a multiple of 8).
__device__ __forceinline__ void stage_panel(const _Float16* __restrict__ src,
                                            _Float16* __restrict__ dst,
                                            int halves, int tid, int nthreads) {
  const uint4* __restrict__ s = (const uint4*)src;
  uint4* d = (uint4*)dst;
  const int vecs = halves >> 3;                  // 8 halves = 16 bytes
  for (int i = tid; i < vecs; i += nthreads) d[i] = s[i];
}

// ---------------------------------------------------------------------------
// Edge message kernel: 16-edge tile per wave, full Dout extent in registers.
//   m = relu( concat_row @ W + b )  -> atomicAdd into hout[dst[e], :]
// Packed f16 B panel staged in LDS once per block (panel is block-uniform:
// the E/2 tile boundary = 25000 is a multiple of the 8 tiles/block).
// ---------------------------------------------------------------------------
template<int KC, int NT>
__global__ void edge_conv_wmma(const float* __restrict__ x,      // [Nnodes, D]
                               const int*   __restrict__ srcIdx, // [E]
                               const int*   __restrict__ dstIdx, // [E]
                               const float* __restrict__ ea,     // [E, ED]
                               const _Float16* __restrict__ Pa, const float* __restrict__ ba,
                               const _Float16* __restrict__ Pb, const float* __restrict__ bb,
                               float* __restrict__ hout,         // [Nnodes, Dout]
                               int D, int Kin, int Dout, int E) {
  __shared__ __align__(32) _Float16 sB[KC * NT * 512];

  const int wavesPerBlk = blockDim.x >> 5;
  const bool secondBlk = ((long)blockIdx.x * wavesPerBlk * 16) >= (long)(E >> 1);

  stage_panel(secondBlk ? Pb : Pa, sB, KC * NT * 512, threadIdx.x, blockDim.x);
  __syncthreads();

  const int wave = (int)((blockIdx.x * blockDim.x + threadIdx.x) >> 5);
  const int lane = threadIdx.x & 31;
  const int tiles = E >> 4;
  if (wave >= tiles) return;

  const long e0 = (long)wave * 16;
  const bool second = (e0 >= (E >> 1));
  const float* __restrict__ bias = second ? bb : ba;

  // A-matrix row for this lane (lanes 0..15 and 16..31 share rows 0..15).
  const int  mrow = lane & 15;
  const long eA   = e0 + mrow;
  const int  se   = srcIdx[eA];
  const int  de   = dstIdx[eA];

  v8f acc[NT] = {};

  // CDNA5 f16 A 16x32 layout: lanes<16 hold K {0..7,16..23}; lanes>=16 {8..15,24..31}
  const int kb0 = (lane < 16) ? 0  : 8;
  const int kb1 = (lane < 16) ? 16 : 24;

  for (int kc = 0; kc < KC; ++kc) {
    v16h a;
#pragma unroll
    for (int j = 0; j < 8; ++j)
      a[j]     = (_Float16)fetch_concat(x, ea, D, Kin, se, de, eA, kc * 32 + kb0 + j, second);
#pragma unroll
    for (int j = 0; j < 8; ++j)
      a[8 + j] = (_Float16)fetch_concat(x, ea, D, Kin, se, de, eA, kc * 32 + kb1 + j, second);

#pragma unroll
    for (int nt = 0; nt < NT; ++nt) {
      const v16h b = *(const v16h*)&sB[(((nt * KC) + kc) * 32 + lane) * 16];
      acc[nt] = __builtin_amdgcn_wmma_f32_16x16x32_f16(
          false, a, false, b, (short)0, acc[nt], false, false);
    }
  }

  // Epilogue: bias + relu + scatter-add to destination nodes.
  const int mb = (lane < 16) ? 0 : 8;
  int dsts[8];
#pragma unroll
  for (int v = 0; v < 8; ++v) dsts[v] = dstIdx[e0 + mb + v];

#pragma unroll
  for (int nt = 0; nt < NT; ++nt) {
    const int n = nt * 16 + (lane & 15);
    if (n >= Dout) continue;
    const float bv = bias[n];
#pragma unroll
    for (int v = 0; v < 8; ++v) {
      float m = fmaxf(acc[nt][v] + bv, 0.0f);
      atomicAdd(&hout[(long)dsts[v] * Dout + n], m);
    }
  }
}

// ---------------------------------------------------------------------------
// Dense GEMM + bias (+optional ReLU): out[M,N] = act(A[M,K] @ Wpacked + b)
// M divisible by 16 (50000/16 = 3125).  Packed B panel staged in LDS.
// ---------------------------------------------------------------------------
template<int KC, int NT, bool RELU>
__global__ void gemm_bias_wmma(const float* __restrict__ A,
                               const _Float16* __restrict__ P,
                               const float* __restrict__ bias,
                               float* __restrict__ out,
                               int M, int K, int Nout) {
  __shared__ __align__(32) _Float16 sB[KC * NT * 512];
  stage_panel(P, sB, KC * NT * 512, threadIdx.x, blockDim.x);
  __syncthreads();

  const int wave = (int)((blockIdx.x * blockDim.x + threadIdx.x) >> 5);
  const int lane = threadIdx.x & 31;
  const int tiles = M >> 4;
  if (wave >= tiles) return;

  const long r0 = (long)wave * 16;
  const int  mrow = lane & 15;
  const float* __restrict__ Arow = A + (r0 + mrow) * K;

  v8f acc[NT] = {};

  const int kb0 = (lane < 16) ? 0  : 8;
  const int kb1 = (lane < 16) ? 16 : 24;

  for (int kc = 0; kc < KC; ++kc) {
    v16h a;
    if (kc * 32 + 32 <= K) {           // fast path: full chunk, no masking
#pragma unroll
      for (int j = 0; j < 8; ++j) a[j]     = (_Float16)Arow[kc * 32 + kb0 + j];
#pragma unroll
      for (int j = 0; j < 8; ++j) a[8 + j] = (_Float16)Arow[kc * 32 + kb1 + j];
    } else {                           // tail chunk: masked
#pragma unroll
      for (int j = 0; j < 8; ++j) {
        const int k = kc * 32 + kb0 + j;
        a[j] = (_Float16)((k < K) ? Arow[k] : 0.0f);
      }
#pragma unroll
      for (int j = 0; j < 8; ++j) {
        const int k = kc * 32 + kb1 + j;
        a[8 + j] = (_Float16)((k < K) ? Arow[k] : 0.0f);
      }
    }

#pragma unroll
    for (int nt = 0; nt < NT; ++nt) {
      const v16h b = *(const v16h*)&sB[(((nt * KC) + kc) * 32 + lane) * 16];
      acc[nt] = __builtin_amdgcn_wmma_f32_16x16x32_f16(
          false, a, false, b, (short)0, acc[nt], false, false);
    }
  }

  const int mb = (lane < 16) ? 0 : 8;
#pragma unroll
  for (int nt = 0; nt < NT; ++nt) {
    const int n = nt * 16 + (lane & 15);
    if (n >= Nout) continue;
    const float bv = bias[n];
#pragma unroll
    for (int v = 0; v < 8; ++v) {
      float o = acc[nt][v] + bv;
      if (RELU) o = fmaxf(o, 0.0f);
      out[(r0 + mb + v) * Nout + n] = o;
    }
  }
}

// Vectorized self-term copy (h = (1+eps)*x, eps=0).  n must be /4.
__global__ void copy_f32x4(const float* __restrict__ in, float* __restrict__ out, long n4) {
  long i = (long)blockIdx.x * blockDim.x + threadIdx.x;
  long stride = (long)gridDim.x * blockDim.x;
  const float4* __restrict__ s = (const float4*)in;
  float4* d = (float4*)out;
  for (; i < n4; i += stride) d[i] = s[i];
}

extern "C" void kernel_launch(void* const* d_in, const int* in_sizes, int n_in,
                              void* d_out, int out_size, void* d_ws, size_t ws_size,
                              hipStream_t stream) {
  const float* x       = (const float*)d_in[0];   // [N, 107]
  const int*   eidx    = (const int*)  d_in[1];   // [2, E]
  const float* eattr   = (const float*)d_in[2];   // [E, 100]
  const float* lin1a_w = (const float*)d_in[3];
  const float* lin1a_b = (const float*)d_in[4];
  const float* lin1b_w = (const float*)d_in[5];
  const float* lin1b_b = (const float*)d_in[6];
  const float* mlp1_w1 = (const float*)d_in[7];
  const float* mlp1_b1 = (const float*)d_in[8];
  const float* mlp1_w2 = (const float*)d_in[9];
  const float* mlp1_b2 = (const float*)d_in[10];
  const float* lin2a_w = (const float*)d_in[11];
  const float* lin2a_b = (const float*)d_in[12];
  const float* lin2b_w = (const float*)d_in[13];
  const float* lin2b_b = (const float*)d_in[14];
  const float* mlp2_w1 = (const float*)d_in[15];
  const float* mlp2_b1 = (const float*)d_in[16];
  const float* mlp2_w2 = (const float*)d_in[17];
  const float* mlp2_b2 = (const float*)d_in[18];
  float* out = (float*)d_out;

  const int* srcIdx = eidx;            // edge_index[0]
  const int* dstIdx = eidx + NEDGES;   // edge_index[1]

  // ---- workspace layout ----
  // floats: h1 [N*107] @0 | t1 [N*100] | g1 [N*100]   (h2 reuses h1, t2 reuses t1)
  // then f16 packed weights (16B aligned: 15,350,000*4 bytes is /16).
  float* ws = (float*)d_ws;
  float* h1 = ws;
  float* t1 = ws + (long)NNODES * D1;
  float* g1 = t1 + (long)NNODES * H1D;
  float* h2 = h1;
  float* t2 = t1;

  _Float16* pk = (_Float16*)(g1 + (long)NNODES * H1D);
  const int SZ_EDGE = 10 * 7 * 512;   // KC=10, NT=7 -> 35840 halves
  const int SZ_47   = 4 * 7 * 512;    // 14336
  const int SZ_44   = 4 * 4 * 512;    // 8192
  const int SZ_27   = 2 * 7 * 512;    // 7168
  _Float16* P_l1a = pk;                 pk += SZ_EDGE;
  _Float16* P_l1b = pk;                 pk += SZ_EDGE;
  _Float16* P_l2a = pk;                 pk += SZ_EDGE;
  _Float16* P_l2b = pk;                 pk += SZ_EDGE;
  _Float16* P_m1a = pk;                 pk += SZ_47;
  _Float16* P_m1b = pk;                 pk += SZ_47;
  _Float16* P_m2a = pk;                 pk += SZ_44;
  _Float16* P_m2b = pk;                 pk += SZ_27;

  // ---- pack weights to f16 B-tile layout ----
  const int K1 = 2 * D1 + ED;   // 314
  const int K2 = 2 * H1D + ED;  // 300
  pack_weight_f16<<<(SZ_EDGE + 255) / 256, 256, 0, stream>>>(lin1a_w, P_l1a, K1, D1, 10, 7);
  pack_weight_f16<<<(SZ_EDGE + 255) / 256, 256, 0, stream>>>(lin1b_w, P_l1b, K1, D1, 10, 7);
  pack_weight_f16<<<(SZ_EDGE + 255) / 256, 256, 0, stream>>>(lin2a_w, P_l2a, K2, H1D, 10, 7);
  pack_weight_f16<<<(SZ_EDGE + 255) / 256, 256, 0, stream>>>(lin2b_w, P_l2b, K2, H1D, 10, 7);
  pack_weight_f16<<<(SZ_47 + 255) / 256, 256, 0, stream>>>(mlp1_w1, P_m1a, D1, H1D, 4, 7);
  pack_weight_f16<<<(SZ_47 + 255) / 256, 256, 0, stream>>>(mlp1_w2, P_m1b, H1D, H1D, 4, 7);
  pack_weight_f16<<<(SZ_44 + 255) / 256, 256, 0, stream>>>(mlp2_w1, P_m2a, H1D, H2D, 4, 4);
  pack_weight_f16<<<(SZ_27 + 255) / 256, 256, 0, stream>>>(mlp2_w2, P_m2b, H2D, DIMO, 2, 7);

  const int edgeTiles = NEDGES / 16;                 // 50000
  const int edgeBlocks = (edgeTiles + 7) / 8;        // 8 waves / 256-thread block
  const int nodeTiles = NNODES / 16;                 // 3125
  const int nodeBlocks = (nodeTiles + 7) / 8;

  // 1) h1 = x (self term)
  {
    long n4 = (long)NNODES * D1 / 4;
    copy_f32x4<<<(int)((n4 + 255) / 256), 256, 0, stream>>>(x, h1, n4);
  }
  // 2) conv1 edge messages -> scatter-add into h1.  Kin=314 (KC=10), Dout=107 (NT=7)
  edge_conv_wmma<10, 7><<<edgeBlocks, 256, 0, stream>>>(
      x, srcIdx, dstIdx, eattr, P_l1a, lin1a_b, P_l1b, lin1b_b,
      h1, D1, K1, D1, NEDGES);
  // 3) t1 = relu(h1 @ mlp1_w1 + b1)   K=107 (KC=4), N=100 (NT=7)
  gemm_bias_wmma<4, 7, true><<<nodeBlocks, 256, 0, stream>>>(
      h1, P_m1a, mlp1_b1, t1, NNODES, D1, H1D);
  // 4) g1 = relu(t1 @ mlp1_w2 + b2)   (mlp output + inter-conv ReLU fused)
  gemm_bias_wmma<4, 7, true><<<nodeBlocks, 256, 0, stream>>>(
      t1, P_m1b, mlp1_b2, g1, NNODES, H1D, H1D);
  // 5) h2 = g1 (self term)
  {
    long n4 = (long)NNODES * H1D / 4;
    copy_f32x4<<<(int)((n4 + 255) / 256), 256, 0, stream>>>(g1, h2, n4);
  }
  // 6) conv2 edge messages -> scatter-add into h2.  Kin=300 (KC=10), Dout=100 (NT=7)
  edge_conv_wmma<10, 7><<<edgeBlocks, 256, 0, stream>>>(
      g1, srcIdx, dstIdx, eattr, P_l2a, lin2a_b, P_l2b, lin2b_b,
      h2, H1D, K2, H1D, NEDGES);
  // 7) t2 = relu(h2 @ mlp2_w1 + b1)   K=100 (KC=4), N=64 (NT=4)
  gemm_bias_wmma<4, 4, true><<<nodeBlocks, 256, 0, stream>>>(
      h2, P_m2a, mlp2_b1, t2, NNODES, H1D, H2D);
  // 8) out = t2 @ mlp2_w2 + b2 (no final ReLU)  K=64 (KC=2), N=100 (NT=7)
  gemm_bias_wmma<2, 7, false><<<nodeBlocks, 256, 0, stream>>>(
      t2, P_m2b, mlp2_b2, out, NNODES, H2D, DIMO);
}
